// GRU_19318762897924
// MI455X (gfx1250) — matrix operand order repaired and checked by reading, not compiled, not run on hardware
//
#include <hip/hip_runtime.h>
#include <hip/hip_bf16.h>
#include <stdint.h>

// GRU: T=512, B=64, D_IN=D_H=1024, fp32 in/out, bf16 WMMA internally.
#define T_STEPS 512
#define BATCH   64
#define DH      1024
#define DIN     1024

typedef __attribute__((ext_vector_type(16))) __bf16 v16bf;
typedef __attribute__((ext_vector_type(8)))  float  v8f;

union BF16Tile { v16bf v; uint4 q[2]; };

__device__ __forceinline__ unsigned short f2bf(float f) {
    unsigned u = __float_as_uint(f);
    unsigned r = u + 0x7FFFu + ((u >> 16) & 1u);   // round-to-nearest-even
    return (unsigned short)(r >> 16);
}

__device__ __forceinline__ float sigm(float x) {
    return 1.0f / (1.0f + __expf(-x));
}

// ---------------------------------------------------------------------------
// Kernel 0a: init h state (fp32 + bf16 mirror) from `state` input, zero barrier
// ---------------------------------------------------------------------------
__global__ void gru_init(const float* __restrict__ state,
                         float* __restrict__ h_state,
                         unsigned short* __restrict__ h_bf,
                         unsigned* __restrict__ cnt) {
    int i = blockIdx.x * blockDim.x + threadIdx.x;
    if (i < BATCH * DH) {
        float v = state[i];
        h_state[i] = v;
        h_bf[i]    = f2bf(v);
    }
    if (i == 0) *cnt = 0u;
}

// ---------------------------------------------------------------------------
// Kernel 0b: fp32 -> bf16 conversion (X)
// ---------------------------------------------------------------------------
__global__ void gru_cvt(const float* __restrict__ src,
                        unsigned short* __restrict__ dst, size_t n) {
    size_t i = (size_t)blockIdx.x * blockDim.x + threadIdx.x;
    size_t stride = (size_t)gridDim.x * blockDim.x;
    for (; i < n; i += stride) dst[i] = f2bf(src[i]);
}

// ---------------------------------------------------------------------------
// Kernel 0c: repack a [K=1024, N=1024] row-major fp32 weight matrix into
// WMMA bf16 B-tile layout: [kt(32)][nt(64)][lane(32)][16] contiguous per lane.
//   lane l holds column nt*16 + (l%16), K = kt*32 + (j<8 ? 8*(l/16) + j : 16 + 8*(l/16) + j - 8)
// ---------------------------------------------------------------------------
__global__ void gru_pack(const float* __restrict__ W,
                         unsigned short* __restrict__ dst) {
    int idx = blockIdx.x * blockDim.x + threadIdx.x;   // 0 .. 1M
    if (idx >= DIN * DH) return;
    int tile   = idx >> 9;        // 512 elements per tile
    int within = idx & 511;
    int lane   = within >> 4;
    int j      = within & 15;
    int kt     = tile >> 6;       // / 64 n-tiles
    int nt     = tile & 63;
    int col    = nt * 16 + (lane & 15);
    int k      = kt * 32 + ((j & 8) << 1) + ((lane >> 4) << 3) + (j & 7);
    dst[idx]   = f2bf(W[(size_t)k * DH + col]);
}

// ---------------------------------------------------------------------------
// Kernel 1: input projections  gates[g] = X @ W_xg + b_g   (fp32 out)
// Block = 256 thr (8 waves). Block tile: 128 rows x 64 cols. Wave: 16 x 64.
// Software-pipelined (double-buffered A/B register tiles).
// grid = (M/128, N/64, 3)
// ---------------------------------------------------------------------------
__launch_bounds__(256)
__global__ void gru_xproj(const unsigned short* __restrict__ Xb,     // [T*B][DIN] bf16
                          const unsigned short* __restrict__ packWx, // 3 packed mats
                          const float* __restrict__ br,
                          const float* __restrict__ bz,
                          const float* __restrict__ bh,
                          float* __restrict__ gates) {               // [3][T*B][DH]
    const int g      = blockIdx.z;
    const int lane   = threadIdx.x & 31;
    const int wave   = threadIdx.x >> 5;
    const int laneN  = lane & 15;
    const int laneHi = lane >> 4;
    const int mrow0  = blockIdx.x * 128 + wave * 16;
    const int ntBase = blockIdx.y * 4;

    const float* bias = (g == 0) ? br : (g == 1) ? bz : bh;

    v8f acc[4];
#pragma unroll
    for (int j = 0; j < 4; ++j) acc[j] = (v8f){0.f,0.f,0.f,0.f,0.f,0.f,0.f,0.f};

    const unsigned short* abase = Xb + (size_t)(mrow0 + laneN) * DIN + laneHi * 8;
    const unsigned short* bmat  = packWx + (size_t)g * (DIN * DH);
    const unsigned short* bbase[4];
#pragma unroll
    for (int j = 0; j < 4; ++j)
        bbase[j] = bmat + ((size_t)(ntBase + j) * 32 + lane) * 16;

    BF16Tile a[2];
    BF16Tile bb[2][4];
    auto loadTiles = [&](int buf, int kt) {
        a[buf].q[0] = *(const uint4*)(abase + kt * 32);
        a[buf].q[1] = *(const uint4*)(abase + kt * 32 + 16);
        size_t koff = (size_t)kt * (64 * 512);
#pragma unroll
        for (int j = 0; j < 4; ++j) {
            bb[buf][j].q[0] = *(const uint4*)(bbase[j] + koff);
            bb[buf][j].q[1] = *(const uint4*)(bbase[j] + koff + 8);
        }
    };

    loadTiles(0, 0);
    for (int kt = 0; kt < 32; ++kt) {
        int cur = kt & 1;
        if (kt < 31) loadTiles(cur ^ 1, kt + 1);   // prefetch next K-slab
#pragma unroll
        for (int j = 0; j < 4; ++j)
            acc[j] = __builtin_amdgcn_wmma_f32_16x16x32_bf16(
                false, a[cur].v, false, bb[cur][j].v, (short)0, acc[j], false, false);
    }

    float* gbase = gates + (size_t)g * ((size_t)T_STEPS * BATCH * DH);
#pragma unroll
    for (int j = 0; j < 4; ++j) {
        int col = (ntBase + j) * 16 + laneN;
        float bv = bias[col];
#pragma unroll
        for (int r = 0; r < 8; ++r) {
            int row = mrow0 + laneHi * 8 + r;
            gbase[(size_t)row * DH + col] = acc[j][r] + bv;
        }
    }
}

// ---------------------------------------------------------------------------
// Kernel 2: persistent recurrent scan. 64 WGs x 256 thr; WG owns 16 columns.
// WG's 96 KB packed-weight slice is staged in (CDNA5 big-)LDS once, then all
// 512 steps read B-tiles via ds_load_b128. A-tiles double-buffered from L2.
// Per step: Phase A (R,Z) -> grid barrier -> Phase B (Hhat, blend) -> barrier.
// Dynamic LDS: 96 KB weights + 4 KB Z + 4 KB partials = 104 KB.
// ---------------------------------------------------------------------------
#define LWH_U16   (3 * 32 * 32 * 16)          // 49152 ushorts = 96 KB
#define SM_BYTES  (LWH_U16 * 2 + 4096 + 4096) // 104 KB

__launch_bounds__(256)
__global__ void gru_scan(const float* __restrict__ gates,            // [3][T*B][DH]
                         const unsigned short* __restrict__ packWh,  // 3 packed mats
                         float* __restrict__ h_state,                // [B][DH] fp32
                         unsigned short* __restrict__ h_bf,          // [B][DH] bf16
                         unsigned short* __restrict__ rh_bf,         // [B][DH] bf16
                         float* __restrict__ out,                    // [T][B][DH] + [B][DH]
                         unsigned* __restrict__ bar_cnt) {
    extern __shared__ char smem[];
    unsigned short* lwh   = (unsigned short*)smem;                   // [g][kt][lane][16]
    float*          z_sm  = (float*)(smem + LWH_U16 * 2);            // [64][16]
    float*          p_sm  = (float*)(smem + LWH_U16 * 2 + 4096);     // [4][32][8]

    const int nt     = blockIdx.x;            // n-tile (16 cols) owned by this WG
    const int n0     = nt * 16;
    const int lane   = threadIdx.x & 31;
    const int wave   = threadIdx.x >> 5;      // 0..7
    const int laneN  = lane & 15;
    const int laneHi = lane >> 4;
    const int m      = wave & 3;              // M-tile (16 rows)
    const int gA     = wave >> 2;             // phase A: 0 = R gate, 1 = Z gate
    const int kh     = wave >> 2;             // phase B: K half
    const unsigned NWG = gridDim.x;
    const size_t GS  = (size_t)T_STEPS * BATCH * DH;   // per-gate stride

    // ---- Stage this WG's weight slice into LDS (once) --------------------
    // 6144 uint4 copies: idx = [g(3)][kt(32)][lane(32)][half(2)]
    for (int idx = threadIdx.x; idx < 6144; idx += 256) {
        int half = idx & 1;
        int ln   = (idx >> 1) & 31;
        int kt   = (idx >> 6) & 31;
        int g    = idx >> 11;
        const unsigned short* src = packWh + (size_t)g * (DH * DH)
                                  + (((size_t)kt * 64 + nt) * 32 + ln) * 16 + half * 8;
        unsigned short* dst = lwh + (((size_t)g * 32 + kt) * 32 + ln) * 16 + half * 8;
        *(uint4*)dst = *(const uint4*)src;
    }
    __syncthreads();

    unsigned barTarget = 0;

#define GRID_SYNC()                                                               \
    do {                                                                          \
        __syncthreads();                                                          \
        barTarget += NWG;                                                         \
        if (threadIdx.x == 0) {                                                   \
            __threadfence();                                                      \
            __hip_atomic_fetch_add(bar_cnt, 1u, __ATOMIC_RELEASE,                 \
                                   __HIP_MEMORY_SCOPE_AGENT);                     \
            while (__hip_atomic_load(bar_cnt, __ATOMIC_ACQUIRE,                   \
                                     __HIP_MEMORY_SCOPE_AGENT) < barTarget) {     \
                __builtin_amdgcn_s_sleep(1);                                      \
            }                                                                     \
        }                                                                         \
        __syncthreads();                                                          \
    } while (0)

    const unsigned short* lbA = lwh + (size_t)gA * (32 * 32 * 16); // R or Z slice
    const unsigned short* lbH = lwh + (size_t)2  * (32 * 32 * 16); // Hhat slice

    const int arow = m * 16 + laneN;
    const int col  = n0 + laneN;

    for (int t = 0; t < T_STEPS; ++t) {
        // Prefetch next timestep's gate slab (global_prefetch_b8) while we compute
        if (t + 1 < T_STEPS && threadIdx.x < 192) {
            int pg  = threadIdx.x >> 6;       // 0..2
            int prw = threadIdx.x & 63;       // row
            __builtin_prefetch(gates + (size_t)pg * GS
                               + ((size_t)(t + 1) * BATCH + prw) * DH + n0, 0, 1);
        }

        // ---------------- Phase A: R = sig(Xr + h Whr), Z = sig(Xz + h Whz) ----
        {
            v8f c = (v8f){0.f,0.f,0.f,0.f,0.f,0.f,0.f,0.f};
            const unsigned short* abase = h_bf + (size_t)arow * DH + laneHi * 8;
            BF16Tile a[2], b;
            a[0].q[0] = *(const uint4*)(abase);
            a[0].q[1] = *(const uint4*)(abase + 16);
            for (int kt = 0; kt < 32; ++kt) {
                int cur = kt & 1;
                if (kt < 31) {                 // prefetch next A slab from L2
                    a[cur ^ 1].q[0] = *(const uint4*)(abase + (kt + 1) * 32);
                    a[cur ^ 1].q[1] = *(const uint4*)(abase + (kt + 1) * 32 + 16);
                }
                const unsigned short* bp = lbA + ((size_t)kt * 32 + lane) * 16;
                b.q[0] = *(const uint4*)(bp);       // ds_load_b128
                b.q[1] = *(const uint4*)(bp + 8);
                c = __builtin_amdgcn_wmma_f32_16x16x32_bf16(
                    false, a[cur].v, false, b.v, (short)0, c, false, false);
            }
            const float* gsrc = gates + (size_t)gA * GS + (size_t)t * BATCH * DH;
            if (gA == 0) {
#pragma unroll
                for (int r = 0; r < 8; ++r) {
                    int row  = m * 16 + laneHi * 8 + r;
                    float pre = c[r] + gsrc[(size_t)row * DH + col];
                    float rg  = sigm(pre);
                    float hv  = h_state[(size_t)row * DH + col];
                    rh_bf[(size_t)row * DH + col] = f2bf(rg * hv);
                }
            } else {
#pragma unroll
                for (int r = 0; r < 8; ++r) {
                    int row  = m * 16 + laneHi * 8 + r;
                    float pre = c[r] + gsrc[(size_t)row * DH + col];
                    z_sm[row * 16 + laneN] = sigm(pre);
                }
            }
        }
        GRID_SYNC();   // all R*h slices written & visible

        // ---------------- Phase B: Hhat = tanh(Xh + (R h) Whh); blend ----------
        {
            v8f c = (v8f){0.f,0.f,0.f,0.f,0.f,0.f,0.f,0.f};
            const unsigned short* abase = rh_bf + (size_t)arow * DH + laneHi * 8;
            const int kt0 = kh * 16;
            BF16Tile a[2], b;
            a[0].q[0] = *(const uint4*)(abase + kt0 * 32);
            a[0].q[1] = *(const uint4*)(abase + kt0 * 32 + 16);
            for (int kk = 0; kk < 16; ++kk) {
                int kt  = kt0 + kk;
                int cur = kk & 1;
                if (kk < 15) {
                    a[cur ^ 1].q[0] = *(const uint4*)(abase + (kt + 1) * 32);
                    a[cur ^ 1].q[1] = *(const uint4*)(abase + (kt + 1) * 32 + 16);
                }
                const unsigned short* bp = lbH + ((size_t)kt * 32 + lane) * 16;
                b.q[0] = *(const uint4*)(bp);
                b.q[1] = *(const uint4*)(bp + 8);
                c = __builtin_amdgcn_wmma_f32_16x16x32_bf16(
                    false, a[cur].v, false, b.v, (short)0, c, false, false);
            }
            if (kh == 1) {
#pragma unroll
                for (int r = 0; r < 8; ++r) p_sm[(m * 32 + lane) * 8 + r] = c[r];
            }
            __syncthreads();
            if (kh == 0) {
                const float* gh = gates + 2 * GS + (size_t)t * BATCH * DH;
#pragma unroll
                for (int r = 0; r < 8; ++r) {
                    int row   = m * 16 + laneHi * 8 + r;
                    float pre = c[r] + p_sm[(m * 32 + lane) * 8 + r]
                              + gh[(size_t)row * DH + col];
                    float hh  = tanhf(pre);
                    float z   = z_sm[row * 16 + laneN];
                    float ho  = h_state[(size_t)row * DH + col];
                    float hn  = z * ho + (1.0f - z) * hh;
                    out[((size_t)t * BATCH + row) * DH + col] = hn;
                    h_state[(size_t)row * DH + col] = hn;
                    h_bf[(size_t)row * DH + col]    = f2bf(hn);
                    if (t == T_STEPS - 1)
                        out[(size_t)T_STEPS * BATCH * DH + (size_t)row * DH + col] = hn;
                }
            }
        }
        GRID_SYNC();   // h updates visible before next step's Phase A
    }
#undef GRID_SYNC
}

// ---------------------------------------------------------------------------
// Launch
// ---------------------------------------------------------------------------
extern "C" void kernel_launch(void* const* d_in, const int* in_sizes, int n_in,
                              void* d_out, int out_size, void* d_ws, size_t ws_size,
                              hipStream_t stream) {
    const float* X   = (const float*)d_in[0];
    const float* st  = (const float*)d_in[1];
    const float* Wxr = (const float*)d_in[2];
    const float* Whr = (const float*)d_in[3];
    const float* br  = (const float*)d_in[4];
    const float* Wxz = (const float*)d_in[5];
    const float* Whz = (const float*)d_in[6];
    const float* bz  = (const float*)d_in[7];
    const float* Wxh = (const float*)d_in[8];
    const float* Whh = (const float*)d_in[9];
    const float* bh  = (const float*)d_in[10];
    float* out = (float*)d_out;

    // Workspace carve-up (256B aligned slabs)
    char*  ws  = (char*)d_ws;
    size_t off = 0;
    auto carve = [&](size_t bytes) -> void* {
        void* p = ws + off;
        off = (off + bytes + 255) & ~(size_t)255;
        return p;
    };
    const size_t GS = (size_t)T_STEPS * BATCH * DH;
    float*          gates   = (float*)         carve(3 * GS * sizeof(float));
    unsigned short* Xb      = (unsigned short*)carve((size_t)T_STEPS * BATCH * DIN * 2);
    unsigned short* packWx  = (unsigned short*)carve((size_t)3 * DIN * DH * 2);
    unsigned short* packWh  = (unsigned short*)carve((size_t)3 * DH * DH * 2);
    float*          h_state = (float*)         carve((size_t)BATCH * DH * sizeof(float));
    unsigned short* h_bf    = (unsigned short*)carve((size_t)BATCH * DH * 2);
    unsigned short* rh_bf   = (unsigned short*)carve((size_t)BATCH * DH * 2);
    unsigned*       cnt     = (unsigned*)      carve(256);
    (void)ws_size; (void)in_sizes; (void)n_in; (void)out_size;

    // Prep
    gru_init<<<(BATCH * DH + 255) / 256, 256, 0, stream>>>(st, h_state, h_bf, cnt);
    gru_cvt<<<2048, 256, 0, stream>>>(X, Xb, (size_t)T_STEPS * BATCH * DIN);
    gru_pack<<<4096, 256, 0, stream>>>(Wxr, packWx + 0 * (size_t)DIN * DH);
    gru_pack<<<4096, 256, 0, stream>>>(Wxz, packWx + 1 * (size_t)DIN * DH);
    gru_pack<<<4096, 256, 0, stream>>>(Wxh, packWx + 2 * (size_t)DIN * DH);
    gru_pack<<<4096, 256, 0, stream>>>(Whr, packWh + 0 * (size_t)DH * DH);
    gru_pack<<<4096, 256, 0, stream>>>(Whz, packWh + 1 * (size_t)DH * DH);
    gru_pack<<<4096, 256, 0, stream>>>(Whh, packWh + 2 * (size_t)DH * DH);

    // Input projections: grid (M/128, N/64, gates)
    dim3 g1((T_STEPS * BATCH) / 128, DH / 64, 3);
    gru_xproj<<<g1, 256, 0, stream>>>(Xb, packWx, br, bz, bh, gates);

    // Persistent recurrent scan: 64 co-resident WGs, one per 16-column slice,
    // 104 KB dynamic LDS (gfx1250 supports 320 KB per workgroup).
    gru_scan<<<DH / 16, 256, SM_BYTES, stream>>>(gates, packWh, h_state, h_bf,
                                                 rh_bf, out, cnt);
}
